// MultiHeadAttentionBlock_395136991379
// MI455X (gfx1250) — matrix operand order repaired and checked by reading, not compiled, not run on hardware
//
#include <hip/hip_runtime.h>
#include <hip/hip_bf16.h>

typedef __bf16 bf16_t;
typedef __attribute__((ext_vector_type(16))) __bf16 v16bf;
typedef __attribute__((ext_vector_type(8)))  float  v8f;

// Problem constants
#define Bc 4
#define Sc 2048
#define Dc 1024
#define Hc 16
#define Kc 64
#define Vc 64
#define HKc (Hc * Kc)   // 1024
#define Mrows (Bc * Sc) // 8192

// GEMM tiling (main GEMM kernel): 128x128 block, 8 waves of 32x64
#define GBM 128
#define GBN 128
#define GBK 32
// AV kernel tiling: 128x64 block, 8 waves of 32x32
#define BM 128
#define BN 64
#define BK 32
#define LDK 48          // padded LDS K-stride (bf16 elems): 96B rows, 16B aligned

// ---------------------------------------------------------------------------
// WMMA fragment loaders (ISA 7.12.2 layouts, wave32)
// A 16x32 bf16: lane m = lane&15, half = lane>>4;
//   elem i: K = (i<8 ? half*8+i : 16+half*8+(i-8))  -> two 16B LDS loads
// ---------------------------------------------------------------------------
__device__ inline v16bf load_frag_a(const bf16_t* base, int row0, int stride) {
    const int lane = threadIdx.x & 31;
    const int hlf  = lane >> 4;
    const bf16_t* p = base + (size_t)(row0 + (lane & 15)) * stride + hlf * 8;
    v16bf f;
#pragma unroll
    for (int i = 0; i < 8; ++i) { f[i] = p[i]; f[8 + i] = p[16 + i]; }
    return f;
}

// B 32x16 bf16 (stored as N rows x K cols in LDS): lane n = lane&15,
// elem i: K = (i<8 ? half*8+i : 16+half*8+(i-8))  -> two 16B LDS loads
__device__ inline v16bf load_frag_b(const bf16_t* base, int col0, int stride) {
    const int lane = threadIdx.x & 31;
    const int hlf  = lane >> 4;
    const bf16_t* p = base + (size_t)(col0 + (lane & 15)) * stride + hlf * 8;
    v16bf f;
#pragma unroll
    for (int i = 0; i < 8; ++i) { f[i] = p[i]; f[8 + i] = p[16 + i]; }
    return f;
}

__device__ inline v8f wmma_bf16(v16bf a, v16bf b, v8f c) {
    return __builtin_amdgcn_wmma_f32_16x16x32_bf16(false, a, false, b,
                                                   (short)0, c, false, false);
}

// ---------------------------------------------------------------------------
// Generic GEMM: C[m,n] = alpha * sum_d A[m,d]*W[n,d] + bias[n]
// A: (M x Kdim) bf16 row-major (lda), W: (N x Kdim) bf16 row-major (ldb)
// Grid: (M/GBM, N/GBN), Block: 256 (8 waves: 4 along M x 2 along N, 32x64 each)
// Double-buffered LDS, register-staged global loads, one barrier per K-step.
// ---------------------------------------------------------------------------
template <typename OutT>
__global__ void __launch_bounds__(256)
gemm_bf16_k(const bf16_t* __restrict__ A, int lda,
            const bf16_t* __restrict__ W, int ldb,
            const float* __restrict__ bias,
            OutT* __restrict__ C, int ldc,
            int Kdim, float alpha) {
    __shared__ __align__(16) bf16_t As[2][GBM * LDK];
    __shared__ __align__(16) bf16_t Bs[2][GBN * LDK];

    const int tid  = threadIdx.x;
    const int wave = tid >> 5;
    const int wm   = wave & 3;   // 4 waves along M (32 rows each)
    const int wn   = wave >> 2;  // 2 waves along N (64 cols each)
    const int m0   = blockIdx.x * GBM;
    const int n0   = blockIdx.y * GBN;

    // Global staging addresses: each thread owns 16 bf16 of A and of B per tile
    const int r  = tid >> 1;
    const int cc = (tid & 1) * 16;
    const bf16_t* gA = A + (size_t)(m0 + r) * lda + cc;
    const bf16_t* gB = W + (size_t)(n0 + r) * ldb + cc;

    v8f acc[2][4] = {};

    // Prime the pipeline: tile 0 -> regs -> LDS buffer 0
    float4 ra0 = ((const float4*)gA)[0];
    float4 ra1 = ((const float4*)(gA + 8))[0];
    float4 rb0 = ((const float4*)gB)[0];
    float4 rb1 = ((const float4*)(gB + 8))[0];
    {
        float4* da = (float4*)(As[0] + r * LDK + cc);
        da[0] = ra0; da[1] = ra1;
        float4* db = (float4*)(Bs[0] + r * LDK + cc);
        db[0] = rb0; db[1] = rb1;
    }

    int cur = 0;
    for (int k0 = 0; k0 < Kdim; k0 += GBK) {
        __syncthreads();   // LDS[cur] visible to all waves
        const bool more = (k0 + GBK) < Kdim;
        if (more) {        // issue next tile's global loads before the WMMAs
            ra0 = ((const float4*)(gA + k0 + GBK))[0];
            ra1 = ((const float4*)(gA + k0 + GBK + 8))[0];
            rb0 = ((const float4*)(gB + k0 + GBK))[0];
            rb1 = ((const float4*)(gB + k0 + GBK + 8))[0];
            __builtin_prefetch(gA + k0 + 2 * GBK, 0, 0);
            __builtin_prefetch(gB + k0 + 2 * GBK, 0, 0);
        }

        v16bf a0 = load_frag_a(As[cur], wm * 32 + 0,  LDK);
        v16bf a1 = load_frag_a(As[cur], wm * 32 + 16, LDK);
        v16bf b0 = load_frag_b(Bs[cur], wn * 64 + 0,  LDK);
        v16bf b1 = load_frag_b(Bs[cur], wn * 64 + 16, LDK);
        v16bf b2 = load_frag_b(Bs[cur], wn * 64 + 32, LDK);
        v16bf b3 = load_frag_b(Bs[cur], wn * 64 + 48, LDK);
        acc[0][0] = wmma_bf16(a0, b0, acc[0][0]);
        acc[0][1] = wmma_bf16(a0, b1, acc[0][1]);
        acc[0][2] = wmma_bf16(a0, b2, acc[0][2]);
        acc[0][3] = wmma_bf16(a0, b3, acc[0][3]);
        acc[1][0] = wmma_bf16(a1, b0, acc[1][0]);
        acc[1][1] = wmma_bf16(a1, b1, acc[1][1]);
        acc[1][2] = wmma_bf16(a1, b2, acc[1][2]);
        acc[1][3] = wmma_bf16(a1, b3, acc[1][3]);

        if (more) {        // stage next tile into the other LDS buffer
            float4* da = (float4*)(As[cur ^ 1] + r * LDK + cc);
            da[0] = ra0; da[1] = ra1;
            float4* db = (float4*)(Bs[cur ^ 1] + r * LDK + cc);
            db[0] = rb0; db[1] = rb1;
        }
        cur ^= 1;
    }

    // Epilogue: C layout — elem i: M = i + 8*half, N = lane&15
    const int lane = tid & 31;
    const int hlf  = lane >> 4;
    const int nl   = lane & 15;
#pragma unroll
    for (int ti = 0; ti < 2; ++ti)
#pragma unroll
        for (int tj = 0; tj < 4; ++tj) {
            const int rowb = m0 + wm * 32 + ti * 16 + 8 * hlf;
            const int ncol = n0 + wn * 64 + tj * 16 + nl;
            const float bv = bias ? bias[ncol] : 0.0f;
#pragma unroll
            for (int i = 0; i < 8; ++i) {
                float v = acc[ti][tj][i] * alpha + bv;
                C[(size_t)(rowb + i) * ldc + ncol] = (OutT)v;
            }
        }
}

// ---------------------------------------------------------------------------
// Column softmax stats over the QUERY axis: per key column s,
// m[s] = max_q scores[q,s]; invz[s] = 1 / sum_q exp(scores[q,s]-m[s])
// ---------------------------------------------------------------------------
__global__ void __launch_bounds__(256)
col_softmax_stats_k(const float* __restrict__ S, float* __restrict__ colm,
                    float* __restrict__ colinvz) {
    const int s = blockIdx.x * 256 + threadIdx.x;
    float m = -3.4e38f;
    for (int q = 0; q < Sc; ++q) m = fmaxf(m, S[(size_t)q * Sc + s]);
    float z = 0.0f;
    for (int q = 0; q < Sc; ++q) z += __expf(S[(size_t)q * Sc + s] - m);
    colm[s]    = m;
    colinvz[s] = 1.0f / z;
}

// ---------------------------------------------------------------------------
// out[q,v] = sum_s exp(scores[q,s]-m[s])*invz[s] * V[s,v]   (WMMA, bf16 P)
// Grid: S/BM blocks. Block tile 128(q) x 64(v), K loop over s in steps of 32.
// ---------------------------------------------------------------------------
__global__ void __launch_bounds__(256)
attn_av_k(const float* __restrict__ S, const float* __restrict__ colm,
          const float* __restrict__ colinvz, const bf16_t* __restrict__ Vv,
          bf16_t* __restrict__ O) {
    __shared__ __align__(16) bf16_t As[BM * LDK];  // P tile (q x s)
    __shared__ __align__(16) bf16_t Bs[Vc * LDK];  // V tile transposed (v x s)

    const int tid  = threadIdx.x;
    const int wave = tid >> 5;
    const int wm   = wave & 3;
    const int wn   = wave >> 2;
    const int m0   = blockIdx.x * BM;

    v8f acc[2][2] = {};

    for (int s0 = 0; s0 < Sc; s0 += BK) {
        {   // P tile: 128x32, softmax weight computed on the fly -> bf16
            const int r = tid >> 1, c = (tid & 1) * 16;
            const float* srow = S + (size_t)(m0 + r) * Sc + s0 + c;
            bf16_t* dst = As + r * LDK + c;
#pragma unroll
            for (int i = 0; i < 16; ++i) {
                const float mm = colm[s0 + c + i];
                const float iz = colinvz[s0 + c + i];
                dst[i] = (bf16_t)(__expf(srow[i] - mm) * iz);
            }
        }
        {   // V tile transpose into LDS: Bs[n][k] = V[s0+k][n]
            const int n = tid & 63, kb = (tid >> 6) * 8;
#pragma unroll
            for (int kk = 0; kk < 8; ++kk)
                Bs[n * LDK + kb + kk] = Vv[(size_t)(s0 + kb + kk) * Vc + n];
        }
        __syncthreads();

        v16bf a0 = load_frag_a(As, wm * 32 + 0,  LDK);
        v16bf a1 = load_frag_a(As, wm * 32 + 16, LDK);
        v16bf b0 = load_frag_b(Bs, wn * 32 + 0,  LDK);
        v16bf b1 = load_frag_b(Bs, wn * 32 + 16, LDK);
        acc[0][0] = wmma_bf16(a0, b0, acc[0][0]);
        acc[0][1] = wmma_bf16(a0, b1, acc[0][1]);
        acc[1][0] = wmma_bf16(a1, b0, acc[1][0]);
        acc[1][1] = wmma_bf16(a1, b1, acc[1][1]);
        __syncthreads();
    }

    const int lane = tid & 31;
    const int hlf  = lane >> 4;
    const int nl   = lane & 15;
#pragma unroll
    for (int ti = 0; ti < 2; ++ti)
#pragma unroll
        for (int tj = 0; tj < 2; ++tj) {
            const int rowb = m0 + wm * 32 + ti * 16 + 8 * hlf;
            const int ncol = wn * 32 + tj * 16 + nl;
#pragma unroll
            for (int i = 0; i < 8; ++i)
                O[(size_t)(rowb + i) * Vc + ncol] = (bf16_t)acc[ti][tj][i];
        }
}

// ---------------------------------------------------------------------------
__global__ void __launch_bounds__(256)
f32_to_bf16_k(const float* __restrict__ in, bf16_t* __restrict__ out, int n) {
    const int i = blockIdx.x * 256 + threadIdx.x;
    if (i < n) out[i] = (bf16_t)in[i];
}

// ---------------------------------------------------------------------------
extern "C" void kernel_launch(void* const* d_in, const int* in_sizes, int n_in,
                              void* d_out, int out_size, void* d_ws, size_t ws_size,
                              hipStream_t stream) {
    const float* x    = (const float*)d_in[0];
    const float* Mq_w = (const float*)d_in[1];
    const float* Mq_b = (const float*)d_in[2];
    const float* Mk_w = (const float*)d_in[3];
    const float* Mk_b = (const float*)d_in[4];
    const float* Mv_w = (const float*)d_in[5];
    const float* Mv_b = (const float*)d_in[6];
    const float* Wo_w = (const float*)d_in[7];
    const float* Wo_b = (const float*)d_in[8];
    float* out = (float*)d_out;

    // Workspace layout (bf16 region then fp32 region) — ~108 MB total
    bf16_t* wsb = (bf16_t*)d_ws;
    size_t off = 0;
    bf16_t* xb = wsb + off; off += (size_t)Mrows * Dc;   // 8192x1024
    bf16_t* wq = wsb + off; off += (size_t)HKc * Dc;
    bf16_t* wk = wsb + off; off += (size_t)HKc * Dc;
    bf16_t* wv = wsb + off; off += (size_t)HKc * Dc;
    bf16_t* wo = wsb + off; off += (size_t)HKc * HKc;
    bf16_t* qb = wsb + off; off += (size_t)Mrows * HKc;
    bf16_t* kb = wsb + off; off += (size_t)Mrows * HKc;
    bf16_t* vb = wsb + off; off += (size_t)Mrows * HKc;
    bf16_t* ob = wsb + off; off += (size_t)Mrows * HKc;  // attention output (bf16)
    float* scores  = (float*)(wsb + off);                 // 2048x2048 fp32, reused per (b,h)
    float* colm    = scores + (size_t)Sc * Sc;
    float* colinvz = colm + Sc;

    // 1) fp32 -> bf16 conversions
    const int nx = Mrows * Dc, nw = HKc * Dc;
    f32_to_bf16_k<<<(nx + 255) / 256, 256, 0, stream>>>(x,    xb, nx);
    f32_to_bf16_k<<<(nw + 255) / 256, 256, 0, stream>>>(Mq_w, wq, nw);
    f32_to_bf16_k<<<(nw + 255) / 256, 256, 0, stream>>>(Mk_w, wk, nw);
    f32_to_bf16_k<<<(nw + 255) / 256, 256, 0, stream>>>(Mv_w, wv, nw);
    f32_to_bf16_k<<<(nw + 255) / 256, 256, 0, stream>>>(Wo_w, wo, nw);

    // 2) Q/K/V projections: (8192x1024) = x @ W^T + b, bf16 out
    dim3 gproj(Mrows / GBM, HKc / GBN);
    gemm_bf16_k<bf16_t><<<gproj, 256, 0, stream>>>(xb, Dc, wq, Dc, Mq_b, qb, HKc, Dc, 1.0f);
    gemm_bf16_k<bf16_t><<<gproj, 256, 0, stream>>>(xb, Dc, wk, Dc, Mk_b, kb, HKc, Dc, 1.0f);
    gemm_bf16_k<bf16_t><<<gproj, 256, 0, stream>>>(xb, Dc, wv, Dc, Mv_b, vb, HKc, Dc, 1.0f);

    // 3) Per-(b,h) attention with query-axis softmax.
    //    The reference's view trick means head bh lives at flat offset bh*S*K.
    for (int bh = 0; bh < Bc * Hc; ++bh) {
        const bf16_t* qh = qb + (size_t)bh * Sc * Kc;
        const bf16_t* kh = kb + (size_t)bh * Sc * Kc;
        const bf16_t* vh = vb + (size_t)bh * Sc * Vc;
        bf16_t*       oh = ob + (size_t)bh * Sc * Vc;

        dim3 gsc(Sc / GBM, Sc / GBN);
        gemm_bf16_k<float><<<gsc, 256, 0, stream>>>(qh, Kc, kh, Kc, nullptr,
                                                    scores, Sc, Kc, 0.125f);
        col_softmax_stats_k<<<Sc / 256, 256, 0, stream>>>(scores, colm, colinvz);
        attn_av_k<<<Sc / BM, 256, 0, stream>>>(scores, colm, colinvz, vh, oh);
    }

    // 4) Output projection: (8192x1024) = attout @ Wo^T + b, fp32 out
    dim3 gout(Mrows / GBM, HKc / GBN);
    gemm_bf16_k<float><<<gout, 256, 0, stream>>>(ob, HKc, wo, HKc, Wo_b, out, HKc, HKc, 1.0f);
}
